// TrajPreAttnAvgLongUser_1786706395396
// MI455X (gfx1250) — matrix operand order repaired and checked by reading, not compiled, not run on hardware
//
#include <hip/hip_runtime.h>
#include <hip/hip_bf16.h>
#include <math.h>

// ---------------------------------------------------------------------------
// Types for CDNA5 WMMA (wave32)
// ---------------------------------------------------------------------------
typedef __attribute__((ext_vector_type(16))) __bf16 v16bf;
typedef __attribute__((ext_vector_type(8)))  float  v8f;

__device__ __forceinline__ unsigned short f32_to_bf16_rne(float f) {
    unsigned int u = __float_as_uint(f);
    unsigned int r = u + 0x7fffu + ((u >> 16) & 1u);
    return (unsigned short)(r >> 16);
}
__device__ __forceinline__ unsigned int pack_bf16x2(float lo, float hi) {
    return (unsigned int)f32_to_bf16_rne(lo) |
           ((unsigned int)f32_to_bf16_rne(hi) << 16);
}

// ---------------------------------------------------------------------------
// Problem constants (from the reference)
// ---------------------------------------------------------------------------
#define S_LEN 2048
#define T_LEN 512
#define LOC_N 100000
#define LE 256
#define TE 32
#define HID 512
#define UE 64
#define NSEG 4096
#define IN_DIM 288           // LE + TE
#define FC_IN 1088           // 2*HID + UE

// ---------------------------------------------------------------------------
// Generic bf16-WMMA GEMM:  C = act(A(MxK) @ B(KxN) [+ bias])
// Block tile 128x128, 8 waves (2x4), wave tile 64x32 (eight 16x16x32 WMMAs).
// A and B tiles staged in LDS in fragment-friendly layouts:
//   As[row][k]  (128 x 40 ushort)  -> A frag = 2x ds_load_b128
//   Bt[col][k]  (128 x 40 ushort)  -> B frag = 2x ds_load_b128
// Staging uses fully-coalesced global loads and packed b128 LDS stores;
// software-pipelined so the next tile's HBM loads overlap the 8 WMMAs.
// transB: B stored (N,K) row-major. act: 0 none, 1 tanh.
// grid = (M/128, ceil(N/128)): x-major dispatch groups the 4 M-blocks that
// share a B column panel so W_fc (435 MB) streams from HBM exactly once.
// ---------------------------------------------------------------------------
#define BM 128
#define BN 128
#define BK 32

__global__ __launch_bounds__(256)
void gemm_bf16_wmma(const float* __restrict__ A, const float* __restrict__ B,
                    const float* __restrict__ bias, float* __restrict__ C,
                    int M, int N, int K, int lda, int ldb, int ldc,
                    int transB, int act)
{
    __shared__ unsigned short As[BM][BK + 8];   // row stride 80 B (16B aligned)
    __shared__ unsigned short Bt[BN][BK + 8];   // column-major B tile

    const int tid  = threadIdx.x;
    const int lane = tid & 31;
    const int wave = tid >> 5;
    const int wm   = wave >> 2;       // 0..1  (M sub-block of 64)
    const int wn   = wave & 3;        // 0..3  (N sub-block of 32)
    const int m0   = blockIdx.x * BM;
    const int n0   = blockIdx.y * BN;

    // per-thread staging coordinates
    const int arow = tid >> 1;                 // A: 0..127
    const int acol = (tid & 1) << 4;           // A: 0 or 16
    const int bn   = tid & 127;                // B: column within tile
    const int bk   = (tid >> 7) << 4;          // B: 0 or 16

    v8f acc[4][2];
#pragma unroll
    for (int i = 0; i < 4; ++i)
#pragma unroll
        for (int j = 0; j < 2; ++j) { v8f z = {}; acc[i][j] = z; }

    float4 ar[4], br[4];

    // ---- tile loader: global -> registers ----
    auto load_tiles = [&](int k0) {
        // A: row m0+arow, cols k0+acol .. +15  (b128 fast path)
        {
            int gm = m0 + arow;
            float* f = reinterpret_cast<float*>(ar);
            if (gm < M && k0 + acol + 16 <= K) {
                const float4* src =
                    reinterpret_cast<const float4*>(A + (size_t)gm * lda + k0 + acol);
#pragma unroll
                for (int i = 0; i < 4; ++i) ar[i] = src[i];
            } else {
#pragma unroll
                for (int i = 0; i < 16; ++i) {
                    int kk = k0 + acol + i;
                    f[i] = (gm < M && kk < K) ? A[(size_t)gm * lda + kk] : 0.f;
                }
            }
        }
        // B: one column (16 K-values) per thread
        {
            int gn = n0 + bn;
            float* f = reinterpret_cast<float*>(br);
            if (!transB) {
                // rows k0+bk..+15 of column gn; a wave's 32 lanes read 32
                // consecutive floats of one B row -> full 128B lines.
                if (gn < N && k0 + bk + 16 <= K) {
#pragma unroll
                    for (int i = 0; i < 16; ++i)
                        f[i] = B[(size_t)(k0 + bk + i) * ldb + gn];
                } else {
#pragma unroll
                    for (int i = 0; i < 16; ++i) {
                        int gk = k0 + bk + i;
                        f[i] = (gn < N && gk < K) ? B[(size_t)gk * ldb + gn] : 0.f;
                    }
                }
            } else {
                // B stored (N,K): K-contiguous -> b128 loads
                if (gn < N && k0 + bk + 16 <= K) {
                    const float4* src =
                        reinterpret_cast<const float4*>(B + (size_t)gn * ldb + k0 + bk);
#pragma unroll
                    for (int i = 0; i < 4; ++i) br[i] = src[i];
                } else {
#pragma unroll
                    for (int i = 0; i < 16; ++i) {
                        int gk = k0 + bk + i;
                        f[i] = (gn < N && gk < K) ? B[(size_t)gn * ldb + gk] : 0.f;
                    }
                }
            }
        }
    };

    // ---- registers -> LDS (bf16 convert, packed b128 stores) ----
    auto store_tiles = [&]() {
        {
            uint4 p0, p1;
            p0.x = pack_bf16x2(ar[0].x, ar[0].y); p0.y = pack_bf16x2(ar[0].z, ar[0].w);
            p0.z = pack_bf16x2(ar[1].x, ar[1].y); p0.w = pack_bf16x2(ar[1].z, ar[1].w);
            p1.x = pack_bf16x2(ar[2].x, ar[2].y); p1.y = pack_bf16x2(ar[2].z, ar[2].w);
            p1.z = pack_bf16x2(ar[3].x, ar[3].y); p1.w = pack_bf16x2(ar[3].z, ar[3].w);
            uint4* dst = reinterpret_cast<uint4*>(&As[arow][acol]);
            dst[0] = p0; dst[1] = p1;
        }
        {
            uint4 p0, p1;
            p0.x = pack_bf16x2(br[0].x, br[0].y); p0.y = pack_bf16x2(br[0].z, br[0].w);
            p0.z = pack_bf16x2(br[1].x, br[1].y); p0.w = pack_bf16x2(br[1].z, br[1].w);
            p1.x = pack_bf16x2(br[2].x, br[2].y); p1.y = pack_bf16x2(br[2].z, br[2].w);
            p1.z = pack_bf16x2(br[3].x, br[3].y); p1.w = pack_bf16x2(br[3].z, br[3].w);
            uint4* dst = reinterpret_cast<uint4*>(&Bt[bn][bk]);
            dst[0] = p0; dst[1] = p1;
        }
    };

    load_tiles(0);

    for (int k0 = 0; k0 < K; k0 += BK) {
        store_tiles();
        __syncthreads();

        if (k0 + BK < K) load_tiles(k0 + BK);   // overlap HBM with WMMAs

        // ---- fragments in contiguous k order (k = 16*(lane/16) + e, same for A and B): all 2x ds_load_b128 ----
        union Frag { v16bf v; unsigned short u[16]; };
        Frag afr[4];
        const int ra    = lane & 15;
        const int kbase = (lane >> 4) << 4;     // 0 or 16  (contiguous k = kbase + e, same as B)
#pragma unroll
        for (int mt = 0; mt < 4; ++mt) {
            int row = wm * 64 + mt * 16 + ra;
#pragma unroll
            for (int e = 0; e < 16; ++e)
                afr[mt].u[e] = As[row][kbase + e];
        }
        const int cbn   = lane & 15;
        const int kboff = (lane >> 4) << 4;     // 0 or 16  (contiguous k = kboff + e, same as A)
#pragma unroll
        for (int nt = 0; nt < 2; ++nt) {
            Frag bfr;
            const unsigned short* bp = &Bt[wn * 32 + nt * 16 + cbn][kboff];
#pragma unroll
            for (int e = 0; e < 16; ++e)
                bfr.u[e] = bp[e];
#pragma unroll
            for (int mt = 0; mt < 4; ++mt)
                acc[mt][nt] = __builtin_amdgcn_wmma_f32_16x16x32_bf16(
                    false, afr[mt].v, false, bfr.v,
                    (short)0, acc[mt][nt], false, false);
        }
        __syncthreads();
    }

    // ---- epilogue: D layout lane->col n, vgpr j -> row j + 8*(lane/16) ----
#pragma unroll
    for (int mt = 0; mt < 4; ++mt) {
#pragma unroll
        for (int nt = 0; nt < 2; ++nt) {
            int n  = n0 + wn * 32 + nt * 16 + (lane & 15);
            int mb = m0 + wm * 64 + mt * 16 + ((lane >> 4) << 3);
            float bv = (bias != nullptr && n < N) ? bias[n] : 0.f;
#pragma unroll
            for (int j = 0; j < 8; ++j) {
                int m = mb + j;
                if (m < M && n < N) {
                    float v = acc[mt][nt][j] + bv;
                    if (act == 1) v = tanhf(v);
                    C[(size_t)m * ldc + n] = v;
                }
            }
        }
    }
}

// ---------------------------------------------------------------------------
// Exclusive prefix sum of hcount -> starts (N=4096, trivial serial scan)
// ---------------------------------------------------------------------------
__global__ void scan_starts_kernel(const int* __restrict__ hcount,
                                   int* __restrict__ starts, int n)
{
    if (blockIdx.x == 0 && threadIdx.x == 0) {
        long s = 0;
        for (int i = 0; i < n; ++i) { starts[i] = (int)s; s += hcount[i]; }
    }
}

// ---------------------------------------------------------------------------
// Segment mean of gathered emb_hloc rows + tim2 gather -> A_hist (4096 x 288)
// ---------------------------------------------------------------------------
__global__ __launch_bounds__(256)
void seg_mean_kernel(const int* __restrict__ hloc, const int* __restrict__ htim,
                     const int* __restrict__ hcount, const int* __restrict__ starts,
                     const float* __restrict__ emb_hloc, const float* __restrict__ emb_htim,
                     float* __restrict__ A_hist)
{
    int n = blockIdx.x;
    int c = threadIdx.x;                 // 0..255
    int start = starts[n];
    int cnt   = hcount[n];
    float s = 0.f;
    for (int j = 0; j < cnt; ++j) {
        int r = hloc[start + j];
        s += emb_hloc[(size_t)r * LE + c];
    }
    A_hist[(size_t)n * IN_DIM + c] = s / (float)cnt;
    if (c < TE) {
        int rt = htim[start];
        A_hist[(size_t)n * IN_DIM + LE + c] = emb_htim[rt * TE + c];
    }
}

// ---------------------------------------------------------------------------
// Build GRU input rows for the tail only (rows 1536..2047) -> A_gru (512x288)
// ---------------------------------------------------------------------------
__global__ __launch_bounds__(288)
void build_gru_A_kernel(const int* __restrict__ loc, const int* __restrict__ tim,
                        const float* __restrict__ emb_loc, const float* __restrict__ emb_tim,
                        float* __restrict__ A_gru)
{
    int i = blockIdx.x;                 // 0..511
    int c = threadIdx.x;                // 0..287
    int s = (S_LEN - T_LEN) + i;
    float v;
    if (c < LE) v = emb_loc[(size_t)loc[s] * LE + c];
    else        v = emb_tim[tim[s] * TE + (c - LE)];
    A_gru[(size_t)i * IN_DIM + c] = v;
}

// ---------------------------------------------------------------------------
// GRU gate epilogue: x (512x1536, already + gru_b[0]) -> tail (512x512)
// Also writes tail into cols [0,512) of the fused FC input matrix (ldc=1088).
// ---------------------------------------------------------------------------
__global__ __launch_bounds__(256)
void gru_gate_kernel(const float* __restrict__ x, const float* __restrict__ gru_b,
                     float* __restrict__ tail, float* __restrict__ Afc)
{
    int idx = blockIdx.x * blockDim.x + threadIdx.x;   // 512*512 total
    int i = idx >> 9;
    int n = idx & 511;
    const float* b1 = gru_b + 3 * HID;                 // gru_b[1]
    float xz = x[(size_t)i * 3 * HID + n];
    float xr = x[(size_t)i * 3 * HID + HID + n];
    float xh = x[(size_t)i * 3 * HID + 2 * HID + n];
    float z  = 1.f / (1.f + __expf(-(xz + b1[n])));
    float r  = 1.f / (1.f + __expf(-(xr + b1[HID + n])));
    float hh = tanhf(xh + r * b1[2 * HID + n]);
    float t  = (1.f - z) * hh;
    tail[idx] = t;
    Afc[(size_t)i * FC_IN + n] = t;
}

// ---------------------------------------------------------------------------
// Broadcast uid embedding into cols [1024,1088) of FC input
// ---------------------------------------------------------------------------
__global__ __launch_bounds__(256)
void fill_uemb_kernel(const int* __restrict__ uid, const float* __restrict__ emb_uid,
                      float* __restrict__ Afc)
{
    int idx = blockIdx.x * blockDim.x + threadIdx.x;   // 512*64 total
    int i = idx >> 6;
    int c = idx & 63;
    Afc[(size_t)i * FC_IN + 2 * HID + c] = emb_uid[(size_t)uid[0] * UE + c];
}

// ---------------------------------------------------------------------------
// Row softmax, in place. ONLINE max+sum (single read pass) then normalize:
// 2 passes over global instead of 3 -- saves ~200 MB of HBM traffic on the
// 512 x 100000 output. Wave32 shuffle merges with (m,s) rescaling.
// ---------------------------------------------------------------------------
__global__ void softmax_rows_kernel(float* __restrict__ data, int cols)
{
    __shared__ float redm[32];
    __shared__ float reds[32];
    __shared__ float bcm, bcs;
    int row  = blockIdx.x;
    float* p = data + (size_t)row * cols;
    int lane = threadIdx.x & 31;
    int w    = threadIdx.x >> 5;
    int nw   = blockDim.x >> 5;

    float m = -3.40282347e38f, s = 0.f;
    for (int c = threadIdx.x; c < cols; c += blockDim.x) {
        float x = p[c];
        if (x > m) { s *= __expf(m - x); m = x; }
        s += __expf(x - m);
    }
    // lane merge: (m,s) <- (max, rescaled sum)
#pragma unroll
    for (int o = 16; o > 0; o >>= 1) {
        float mo = __shfl_xor(m, o, 32);
        float so = __shfl_xor(s, o, 32);
        float mn = fmaxf(m, mo);
        s = s * __expf(m - mn) + so * __expf(mo - mn);
        m = mn;
    }
    if (lane == 0) { redm[w] = m; reds[w] = s; }
    __syncthreads();
    if (w == 0) {
        float mm = (lane < nw) ? redm[lane] : -3.40282347e38f;
        float ss = (lane < nw) ? reds[lane] : 0.f;
#pragma unroll
        for (int o = 16; o > 0; o >>= 1) {
            float mo = __shfl_xor(mm, o, 32);
            float so = __shfl_xor(ss, o, 32);
            float mn = fmaxf(mm, mo);
            ss = ss * __expf(mm - mn) + so * __expf(mo - mn);
            mm = mn;
        }
        if (lane == 0) { bcm = mm; bcs = ss; }
    }
    __syncthreads();
    float M  = bcm;
    float inv = 1.f / bcs;
    for (int c = threadIdx.x; c < cols; c += blockDim.x)
        p[c] = __expf(p[c] - M) * inv;
}

// ---------------------------------------------------------------------------
// Host-side orchestration
// ---------------------------------------------------------------------------
extern "C" void kernel_launch(void* const* d_in, const int* in_sizes, int n_in,
                              void* d_out, int out_size, void* d_ws, size_t ws_size,
                              hipStream_t stream)
{
    const int*   loc      = (const int*)  d_in[0];
    const int*   tim      = (const int*)  d_in[1];
    const int*   uid      = (const int*)  d_in[2];
    const int*   hloc     = (const int*)  d_in[3];
    const int*   htim     = (const int*)  d_in[4];
    const int*   hcount   = (const int*)  d_in[5];
    /* d_in[6] target_len == 512, constant */
    const float* emb_loc  = (const float*)d_in[7];
    const float* emb_tim  = (const float*)d_in[8];
    const float* emb_uid  = (const float*)d_in[9];
    const float* emb_hloc = (const float*)d_in[10];
    const float* emb_htim = (const float*)d_in[11];
    const float* gru_W    = (const float*)d_in[12];
    /* d_in[13] gru_U unused by the reference */
    const float* gru_b    = (const float*)d_in[14];
    const float* W_hist   = (const float*)d_in[15];
    const float* b_hist   = (const float*)d_in[16];
    const float* W_fc     = (const float*)d_in[17];
    const float* b_fc     = (const float*)d_in[18];
    float* out = (float*)d_out;

    // workspace carve-up (~28.5 MB), all chunks 16B-aligned
    char*  w       = (char*)d_ws;
    int*   starts  = (int*)w;
    float* A_hist  = (float*)(w + 16384);
    float* hist_in = A_hist  + (size_t)NSEG * IN_DIM;
    float* A_gru   = hist_in + (size_t)NSEG * HID;
    float* x_gru   = A_gru   + (size_t)T_LEN * IN_DIM;
    float* tailb   = x_gru   + (size_t)T_LEN * 3 * HID;
    float* Sc      = tailb   + (size_t)T_LEN * HID;
    float* Afc     = Sc      + (size_t)T_LEN * NSEG;

    // 1) segment starts
    scan_starts_kernel<<<1, 32, 0, stream>>>(hcount, starts, NSEG);

    // 2) history segment-mean + tim gather -> A_hist (4096 x 288)
    seg_mean_kernel<<<NSEG, 256, 0, stream>>>(hloc, htim, hcount, starts,
                                              emb_hloc, emb_htim, A_hist);

    // 3) hist_in = tanh(A_hist @ W_hist + b_hist)   (4096 x 512)
    gemm_bf16_wmma<<<dim3(NSEG / BM, HID / BN), 256, 0, stream>>>(
        A_hist, W_hist, b_hist, hist_in,
        NSEG, HID, IN_DIM, IN_DIM, HID, HID, 0, 1);

    // 4) tail GRU input rows (512 x 288)
    build_gru_A_kernel<<<T_LEN, IN_DIM, 0, stream>>>(loc, tim, emb_loc, emb_tim, A_gru);

    // 5) x = A_gru @ gru_W + gru_b[0]   (512 x 1536)
    gemm_bf16_wmma<<<dim3(T_LEN / BM, 3 * HID / BN), 256, 0, stream>>>(
        A_gru, gru_W, gru_b, x_gru,
        T_LEN, 3 * HID, IN_DIM, IN_DIM, 3 * HID, 3 * HID, 0, 0);

    // 6) gates -> tail; also write FC-input cols [0,512)
    gru_gate_kernel<<<(T_LEN * HID) / 256, 256, 0, stream>>>(x_gru, gru_b, tailb, Afc);

    // 7) scores = tail @ hist_in^T   (512 x 4096)
    gemm_bf16_wmma<<<dim3(T_LEN / BM, NSEG / BN), 256, 0, stream>>>(
        tailb, hist_in, nullptr, Sc,
        T_LEN, NSEG, HID, HID, HID, NSEG, 1, 0);

    // 8) softmax rows of scores
    softmax_rows_kernel<<<T_LEN, 256, 0, stream>>>(Sc, NSEG);

    // 9) context = scores @ hist_in  -> FC-input cols [512,1024)  (ldc trick)
    gemm_bf16_wmma<<<dim3(T_LEN / BM, HID / BN), 256, 0, stream>>>(
        Sc, hist_in, nullptr, Afc + HID,
        T_LEN, HID, NSEG, NSEG, HID, FC_IN, 0, 0);

    // 10) uid embedding -> FC-input cols [1024,1088)
    fill_uemb_kernel<<<(T_LEN * UE) / 256, 256, 0, stream>>>(uid, emb_uid, Afc);

    // 11) logits = Afc @ W_fc + b_fc   (512 x 100000), W_fc streamed once
    gemm_bf16_wmma<<<dim3(T_LEN / BM, (LOC_N + BN - 1) / BN), 256, 0, stream>>>(
        Afc, W_fc, b_fc, out,
        T_LEN, LOC_N, FC_IN, FC_IN, LOC_N, LOC_N, 0, 0);

    // 12) final softmax over 100000 classes, in place in d_out
    softmax_rows_kernel<<<T_LEN, 1024, 0, stream>>>(out, LOC_N);
}
